// SparseMHA_39032662786146
// MI455X (gfx1250) — matrix operand order repaired and checked by reading, not compiled, not run on hardware
//
#include <hip/hip_runtime.h>

// ---------------------------------------------------------------------------
// SparseMHA for MI455X (gfx1250, wave32).
//  Phase 1: 4 dense GEMMs via V_WMMA_F32_16X16X32_BF16 (bf16 in, f32 acc),
//           B staged in LDS by the Tensor Data Mover (tensor_load_to_lds)
//           with TDM row padding for bank-conflict-free ds_load_b128,
//           A fragments double-buffered in registers. B fragments are read
//           with inline-asm ds_load_b128 (opaque: the compiler cannot prove
//           the TDM wrote LDS, so ordinary loads get folded to undef).
//  Phase 2: edge logits, CSR build, per-(node,head) wave softmax+aggregate.
// ---------------------------------------------------------------------------

typedef __attribute__((ext_vector_type(16))) __bf16    v16bf;
typedef __attribute__((ext_vector_type(8)))  float     v8f;
typedef __attribute__((ext_vector_type(4)))  unsigned  v4u;
typedef __attribute__((ext_vector_type(8)))  int       v8i;
typedef __attribute__((ext_vector_type(4)))  int       v4i;

union ABfrag { v4u u[2]; v16bf v; };

#define NNODES 32768
#define NEDGES 262144
#define DMODEL 512
#define NHEAD  8
#define DHEAD  64
#define LDSROW 520   // elements per staged B row: 512 + 8 bf16 pad (1040 B)

__device__ __forceinline__ unsigned short f2bf(float f) {
    unsigned u = __float_as_uint(f);
    u += 0x7FFFu + ((u >> 16) & 1u);   // round-to-nearest-even
    return (unsigned short)(u >> 16);
}

// ---- convert fp32 -> bf16, 4 elements / thread ----------------------------
__global__ __launch_bounds__(256) void cvt_bf16_kernel(
    const float* __restrict__ in, unsigned short* __restrict__ out, int n4) {
    int i = blockIdx.x * 256 + threadIdx.x;
    if (i >= n4) return;
    float4 f = reinterpret_cast<const float4*>(in)[i];
    uint2 p;
    p.x = (unsigned)f2bf(f.x) | ((unsigned)f2bf(f.y) << 16);
    p.y = (unsigned)f2bf(f.z) | ((unsigned)f2bf(f.w) << 16);
    reinterpret_cast<uint2*>(out)[i] = p;
}

// ---- convert + transpose weight [K,N] fp32 -> [N,K] bf16 ------------------
__global__ __launch_bounds__(256) void cvt_wT_kernel(
    const float* __restrict__ W, unsigned short* __restrict__ Wt) {
    int i = blockIdx.x * 256 + threadIdx.x;       // over 512*512
    int k = i >> 9, n = i & 511;
    Wt[(size_t)n * DMODEL + k] = f2bf(W[i]);
}

// ---- TDM: pull a contiguous 64KB B slab into LDS with 16B/row padding -----
// D# per CDNA5 ISA ch8: group0 = {count/flags, lds_addr, global_addr, type},
// group1 = {mask/data_size/pad, tensor dims, tile dims, strides}.
// 1-D transfer: data_size=8B, tensor_dim0=tile_dim0=8192 units (64KB),
// pad_enable, pad_interval code 7 (256 dwords = 1024B), pad_amount code 3
// (4 dwords = 16B)  ->  LDS row stride 1040B (bank-conflict-free fragments).
__device__ __forceinline__ void tdm_load_b_tile(const unsigned short* gsrc) {
    unsigned long long ga = (unsigned long long)(uintptr_t)gsrc;
    v4u g0;
    g0.x = 1u;                                          // count=1
    g0.y = 0u;                                          // lds_addr = 0
    g0.z = (unsigned)(ga & 0xFFFFFFFFu);                // global_addr[31:0]
    g0.w = (unsigned)((ga >> 32) & 0x1FFFFFFu)          // global_addr[56:32]
           | (2u << 30);                                // type = 2 ("image")
    v8i g1;
    g1[0] = (int)((3u << 16)      // data_size = 8B
                | (1u << 20)      // pad_enable
                | (7u << 22)      // pad_interval: 256 dwords
                | (3u << 25));    // pad_amount:   4 dwords (16B)
    g1[1] = (int)(0x2000u << 16); // tensor_dim0 = 8192 (8B units) in [31:16]
    g1[2] = (int)(1u << 16);      // tensor_dim1 = 1
    g1[3] = (int)(0x2000u << 16); // tile_dim0 = 8192
    g1[4] = 0;                    // tile_dim1 = 0 (unused), tile_dim2 = 0
    g1[5] = 8192;                 // tensor_dim0_stride
    g1[6] = 0;
    g1[7] = 0;
    v4i gz4 = {0, 0, 0, 0};
    v8i gz8 = {0, 0, 0, 0, 0, 0, 0, 0};
    __builtin_amdgcn_tensor_load_to_lds(g0, g1, gz4, gz4, gz8, 0);
}

// ---- opaque LDS fragment load: 2 x ds_load_b128 + wait --------------------
__device__ __forceinline__ void lds_load_frag(ABfrag& b, unsigned addr) {
    asm volatile("ds_load_b128 %0, %2\n\t"
                 "ds_load_b128 %1, %2 offset:16\n\t"
                 "s_wait_dscnt 0x0"
                 : "=v"(b.u[0]), "=v"(b.u[1])
                 : "v"(addr));
}

// ---- WMMA GEMM: C[M,512] = A[M,512](bf16) * B(bf16, stored transposed) ----
// block = 256 threads = 8 waves; wave -> one 16-row strip, 4 x 16-col tiles.
__global__ __launch_bounds__(256) void gemm_wmma_kernel(
    const unsigned short* __restrict__ A,    // M x 512 bf16 row-major
    const unsigned short* __restrict__ Bt,   // 512 x 512 bf16, Bt[n][k] = B[k][n]
    const float* __restrict__ bias,          // [512]
    float scale,
    float* __restrict__ Cf,                  // fp32 out (or null)
    unsigned short* __restrict__ Cb) {       // bf16 out (or null)
    __shared__ unsigned short smem[64 * LDSROW];   // 66560 B, LDS offset 0

    const int lane   = threadIdx.x & 31;
    const int wave   = threadIdx.x >> 5;
    const int laneM  = lane & 15;
    const int hi     = lane >> 4;            // 0: lanes 0-15, 1: lanes 16-31
    const int m0     = (blockIdx.x * 8 + wave) * 16;
    const int n0     = blockIdx.y * 64;

    // One wave DMAs the block's 64-column B slab into LDS (TENSORcnt-tracked).
    if (threadIdx.x == 0) {
        tdm_load_b_tile(Bt + (size_t)n0 * DMODEL);
        __builtin_amdgcn_s_wait_tensorcnt(0);
    }
    asm volatile("" ::: "memory");           // LDS was mutated behind LLVM's back
    __syncthreads();

    // 32-bit LDS byte address of smem (generic-pointer low bits = LDS offset)
    const unsigned sbase = (unsigned)(size_t)(const void*)&smem[0];

    v8f acc[4] = {};
    const unsigned short* arow = A + (size_t)(m0 + laneM) * DMODEL;

    // A fragment 16x32: lanes 0-15 hold K 0..15, lanes 16-31 K 16..31
    ABfrag a[2];
    {
        const int ka = hi * 16;
        a[0].u[0] = *reinterpret_cast<const v4u*>(arow + ka);
        a[0].u[1] = *reinterpret_cast<const v4u*>(arow + ka + 8);
    }
#pragma unroll
    for (int k0 = 0; k0 < DMODEL; k0 += 32) {
        const int cur = (k0 >> 5) & 1;
        if (k0 + 32 < DMODEL) {              // prefetch next A fragment
            const int ka = k0 + 32 + hi * 16;
            a[cur ^ 1].u[0] = *reinterpret_cast<const v4u*>(arow + ka);
            a[cur ^ 1].u[1] = *reinterpret_cast<const v4u*>(arow + ka + 8);
        }
#pragma unroll
        for (int t = 0; t < 4; ++t) {
            // B fragment 32x16 from LDS: lanes 0-15 K k0..k0+15, 16-31 k0+16..31
            ABfrag b;
            const unsigned baddr = sbase +
                (unsigned)(((t * 16 + laneM) * LDSROW + k0 + hi * 16) * 2);
            lds_load_frag(b, baddr);
            acc[t] = __builtin_amdgcn_wmma_f32_16x16x32_bf16(
                false, a[cur].v, false, b.v, (short)0, acc[t], false, false);
        }
    }

    // C layout: VGPR r -> row m0 + r + 8*hi, col n0 + 16*t + laneM
#pragma unroll
    for (int t = 0; t < 4; ++t) {
        const int n = n0 + t * 16 + laneM;
        const float bv = bias[n];
#pragma unroll
        for (int r = 0; r < 8; ++r) {
            const int m = m0 + r + hi * 8;
            const float val = (acc[t][r] + bv) * scale;
            if (Cf) Cf[(size_t)m * DMODEL + n] = val;
            else    Cb[(size_t)m * DMODEL + n] = f2bf(val);
        }
    }
}

// ---- per-(edge,head) logits: dot of 64 floats -----------------------------
__global__ __launch_bounds__(256) void edge_logits_kernel(
    const float* __restrict__ q, const float* __restrict__ k,
    const int* __restrict__ row, const int* __restrict__ col,
    float* __restrict__ logits) {
    int t = blockIdx.x * 256 + threadIdx.x;          // E*H threads
    int e = t >> 3, h = t & 7;
    const float4* qp = reinterpret_cast<const float4*>(
        q + (size_t)row[e] * DMODEL + h * DHEAD);
    const float4* kp = reinterpret_cast<const float4*>(
        k + (size_t)col[e] * DMODEL + h * DHEAD);
    float acc = 0.f;
#pragma unroll
    for (int i = 0; i < 16; ++i) {
        float4 a = qp[i], b = kp[i];
        acc += a.x * b.x + a.y * b.y + a.z * b.z + a.w * b.w;
    }
    logits[t] = acc;
}

// ---- CSR build ------------------------------------------------------------
__global__ __launch_bounds__(256) void zero2_kernel(int* a, int* b, int n) {
    int i = blockIdx.x * 256 + threadIdx.x;
    if (i < n) { a[i] = 0; b[i] = 0; }
}

__global__ __launch_bounds__(256) void count_kernel(
    const int* __restrict__ row, int* __restrict__ cnt) {
    int e = blockIdx.x * 256 + threadIdx.x;
    if (e < NEDGES) atomicAdd(&cnt[row[e]], 1);
}

__global__ __launch_bounds__(1024) void scan_kernel(
    const int* __restrict__ cnt, int* __restrict__ rowptr) {
    __shared__ int part[1024];
    const int t = threadIdx.x;
    const int base = t * 32;                         // 1024 threads * 32 = 32768
    int s = 0;
    for (int i = 0; i < 32; ++i) s += cnt[base + i];
    part[t] = s;
    __syncthreads();
    for (int off = 1; off < 1024; off <<= 1) {
        int val = (t >= off) ? part[t - off] : 0;
        __syncthreads();
        part[t] += val;
        __syncthreads();
    }
    int run = (t == 0) ? 0 : part[t - 1];
    for (int i = 0; i < 32; ++i) { rowptr[base + i] = run; run += cnt[base + i]; }
    if (t == 0) rowptr[NNODES] = NEDGES;
}

__global__ __launch_bounds__(256) void scatter_kernel(
    const int* __restrict__ row, const int* __restrict__ rowptr,
    int* __restrict__ fill, int* __restrict__ eidx) {
    int e = blockIdx.x * 256 + threadIdx.x;
    if (e >= NEDGES) return;
    int r = row[e];
    int p = atomicAdd(&fill[r], 1);
    eidx[rowptr[r] + p] = e;
}

// ---- softmax + aggregation: one wave per (node, head) ---------------------
// Edges iterated in increasing edge-id order (min-scan) so the float sum
// order is deterministic regardless of atomic scatter order.
__global__ __launch_bounds__(256) void aggregate_kernel(
    const float* __restrict__ logits, const float* __restrict__ v,
    const int* __restrict__ col, const int* __restrict__ rowptr,
    const int* __restrict__ eidx, float* __restrict__ out) {
    const int w    = blockIdx.x * 8 + (threadIdx.x >> 5);   // N*H waves
    const int lane = threadIdx.x & 31;
    if (w >= NNODES * NHEAD) return;
    const int n = w >> 3, h = w & 7;
    const int beg = rowptr[n], end = rowptr[n + 1];

    float m = -3.402823466e38f;
    for (int i = beg; i < end; ++i)
        m = fmaxf(m, logits[(size_t)eidx[i] * NHEAD + h]);

    float sum = 0.f, a0 = 0.f, a1 = 0.f;
    int last = -1;
    for (int c = beg; c < end; ++c) {
        int cur = 0x7FFFFFFF;
        for (int i = beg; i < end; ++i) {           // deterministic order
            int e = eidx[i];
            if (e > last && e < cur) cur = e;
        }
        last = cur;
        float wgt = __expf(logits[(size_t)cur * NHEAD + h] - m);
        sum += wgt;
        const float* vp = v + (size_t)col[cur] * DMODEL + h * DHEAD;
        a0 += wgt * vp[lane];
        a1 += wgt * vp[lane + 32];
    }
    const float inv = (end > beg) ? 1.0f / sum : 0.0f;
    out[(size_t)n * DMODEL + h * DHEAD + lane]      = a0 * inv;
    out[(size_t)n * DMODEL + h * DHEAD + 32 + lane] = a1 * inv;
}

// ---------------------------------------------------------------------------
extern "C" void kernel_launch(void* const* d_in, const int* in_sizes, int n_in,
                              void* d_out, int out_size, void* d_ws, size_t ws_size,
                              hipStream_t stream) {
    const float* h    = (const float*)d_in[0];
    const int*   row  = (const int*)d_in[1];
    const int*   col  = (const int*)d_in[2];
    const float* W_in = (const float*)d_in[3];
    const float* b_in = (const float*)d_in[4];
    const float* W_q  = (const float*)d_in[5];
    const float* b_q  = (const float*)d_in[6];
    const float* W_k  = (const float*)d_in[7];
    const float* b_k  = (const float*)d_in[8];
    const float* W_v  = (const float*)d_in[9];
    const float* b_v  = (const float*)d_in[10];
    float* out = (float*)d_out;

    // Workspace layout (~280 MB, 256B aligned slices).
    char* ws = (char*)d_ws;
    size_t off = 0;
    auto alloc = [&](size_t bytes) {
        char* p = ws + off;
        off = (off + bytes + 255) & ~(size_t)255;
        return p;
    };
    unsigned short* hb    = (unsigned short*)alloc((size_t)NNODES * DMODEL * 2);
    unsigned short* xb    = (unsigned short*)alloc((size_t)NNODES * DMODEL * 2);
    unsigned short* wtIn  = (unsigned short*)alloc((size_t)DMODEL * DMODEL * 2);
    unsigned short* wtQ   = (unsigned short*)alloc((size_t)DMODEL * DMODEL * 2);
    unsigned short* wtK   = (unsigned short*)alloc((size_t)DMODEL * DMODEL * 2);
    unsigned short* wtV   = (unsigned short*)alloc((size_t)DMODEL * DMODEL * 2);
    float* qf             = (float*)alloc((size_t)NNODES * DMODEL * 4);
    float* kf             = (float*)alloc((size_t)NNODES * DMODEL * 4);
    float* vf             = (float*)alloc((size_t)NNODES * DMODEL * 4);
    float* logits         = (float*)alloc((size_t)NEDGES * NHEAD * 4);
    int*   cnt            = (int*)alloc((size_t)NNODES * 4);
    int*   fill           = (int*)alloc((size_t)NNODES * 4);
    int*   rowptr         = (int*)alloc((size_t)(NNODES + 1) * 4);
    int*   eidx           = (int*)alloc((size_t)NEDGES * 4);
    (void)ws_size; (void)n_in; (void)in_sizes; (void)out_size;

    // 1) precision conversion (bf16) + weight transposes
    cvt_bf16_kernel<<<(NNODES * DMODEL / 4 + 255) / 256, 256, 0, stream>>>(
        h, hb, NNODES * DMODEL / 4);
    cvt_wT_kernel<<<DMODEL * DMODEL / 256, 256, 0, stream>>>(W_in, wtIn);
    cvt_wT_kernel<<<DMODEL * DMODEL / 256, 256, 0, stream>>>(W_q,  wtQ);
    cvt_wT_kernel<<<DMODEL * DMODEL / 256, 256, 0, stream>>>(W_k,  wtK);
    cvt_wT_kernel<<<DMODEL * DMODEL / 256, 256, 0, stream>>>(W_v,  wtV);

    // 2) x = h @ W_in + b_in  (bf16 out), then q/k/v (fp32 out, q scaled)
    dim3 ggrid(NNODES / 128, DMODEL / 64);
    gemm_wmma_kernel<<<ggrid, 256, 0, stream>>>(hb, wtIn, b_in, 1.0f, nullptr, xb);
    gemm_wmma_kernel<<<ggrid, 256, 0, stream>>>(xb, wtQ, b_q, 0.125f, qf, nullptr);
    gemm_wmma_kernel<<<ggrid, 256, 0, stream>>>(xb, wtK, b_k, 1.0f,   kf, nullptr);
    gemm_wmma_kernel<<<ggrid, 256, 0, stream>>>(xb, wtV, b_v, 1.0f,   vf, nullptr);

    // 3) sparse phase
    edge_logits_kernel<<<NEDGES * NHEAD / 256, 256, 0, stream>>>(qf, kf, row, col, logits);
    zero2_kernel<<<NNODES / 256, 256, 0, stream>>>(cnt, fill, NNODES);
    count_kernel<<<NEDGES / 256, 256, 0, stream>>>(row, cnt);
    scan_kernel<<<1, 1024, 0, stream>>>(cnt, rowptr);
    scatter_kernel<<<NEDGES / 256, 256, 0, stream>>>(row, rowptr, fill, eidx);
    aggregate_kernel<<<NNODES * NHEAD / 8, 256, 0, stream>>>(
        logits, vf, col, rowptr, eidx, out);
}
